// SAE_29652454212340
// MI455X (gfx1250) — matrix operand order repaired and checked by reading, not compiled, not run on hardware
//
#include <hip/hip_runtime.h>
#include <hip/hip_bf16.h>

// ---------------------------------------------------------------------------
// TopK Sparse Autoencoder forward for MI455X (gfx1250, wave32, WMMA).
//   B=4096 tokens, D=1024 dims, L=32768 latents, TOPK=32.
// d_out layout: latents[B,L] | out[B,D] | preact[B,L] | num_dead
// ---------------------------------------------------------------------------

constexpr int B_TOK = 4096;
constexpr int D_IN  = 1024;
constexpr int L_LAT = 32768;
constexpr int TOPK  = 32;

// Encode tiling: block = 256 threads = 8 waves arranged 2(M) x 4(N).
// Block tile 64(M) x 256(N), K-slab 32. Each wave: 32x64 = 8 WMMAs/slab.
constexpr int TM = 64;
constexpr int TN = 256;
constexpr int TK = 32;
constexpr int LDS_K = TK + 8;   // bf16 elems per row; 80B rows, dword aligned

typedef __bf16 v16bf __attribute__((ext_vector_type(16)));
typedef float  v8f   __attribute__((ext_vector_type(8)));

__device__ __forceinline__ unsigned pack_bf16(float a, float b) {
    union { __bf16 h[2]; unsigned u; } p;
    p.h[0] = (__bf16)a; p.h[1] = (__bf16)b;   // -> v_cvt_pk_bf16_f32
    return p.u;
}

// ---------------------------------------------------------------------------
// Kernel 1: encode GEMM  preact = x @ W_enc + b_enc   (bf16 WMMA, f32 accum)
// ---------------------------------------------------------------------------
__global__ __launch_bounds__(256) void sae_encode_wmma(
    const float* __restrict__ x, const float* __restrict__ W_enc,
    const float* __restrict__ b_enc, float* __restrict__ preact)
{
    __shared__ __bf16 As[TM][LDS_K];   // x tile, row-major [m][k]
    __shared__ __bf16 Bs[TN][LDS_K];   // W_enc tile, TRANSPOSED [n][k]

    const int tid  = threadIdx.x;
    const int lane = tid & 31;
    const int wave = tid >> 5;
    const int wm   = wave >> 2;        // 0..1 : wave row   (32 M rows each)
    const int wn   = wave & 3;         // 0..3 : wave col   (64 N cols each)
    const int m0   = blockIdx.y * TM;
    const int n0   = blockIdx.x * TN;
    const int half = lane >> 4;        // lane group 0/1
    const int l15  = lane & 15;

    v8f acc[2][4];
    #pragma unroll
    for (int mt = 0; mt < 2; ++mt)
        #pragma unroll
        for (int nt = 0; nt < 4; ++nt)
            #pragma unroll
            for (int r = 0; r < 8; ++r) acc[mt][nt][r] = 0.0f;

    for (int k0 = 0; k0 < D_IN; k0 += TK) {
        // ---- stage x tile: 64x32 -> 1024 packed bf16x2 dwords, 4 per thread
        #pragma unroll
        for (int i = 0; i < 4; ++i) {
            int lin = tid + i * 256;        // pair index
            int m   = lin >> 4;             // 0..63
            int kp  = lin & 15;             // k-pair 0..15
            const float* gp = x + (size_t)(m0 + m) * D_IN + k0 + 2 * kp;
            ((unsigned*)&As[m][0])[kp] = pack_bf16(gp[0], gp[1]);
        }
        // ---- stage W_enc tile: 32x256 -> 4096 packed dwords, 16 per thread
        //      two k-adjacent rows read (each coalesced over n), packed on k
        #pragma unroll
        for (int i = 0; i < 16; ++i) {
            int lin = tid + i * 256;        // pair index
            int n   = lin & 255;            // 0..255
            int kp  = lin >> 8;             // k-pair 0..15
            const float* gp = W_enc + (size_t)(k0 + 2 * kp) * L_LAT + n0 + n;
            ((unsigned*)&Bs[n][0])[kp] = pack_bf16(gp[0], gp[L_LAT]);
        }
        __syncthreads();

        // ---- A fragments (16x32 bf16, contiguous k = 16*half + j), 2 M-tiles
        v16bf af[2];
        #pragma unroll
        for (int mt = 0; mt < 2; ++mt) {
            const int mloc = wm * 32 + mt * 16 + l15;
            #pragma unroll
            for (int j = 0; j < 8; ++j) af[mt][j]     = As[mloc][half * 16 + j];
            #pragma unroll
            for (int j = 0; j < 8; ++j) af[mt][8 + j] = As[mloc][half * 16 + 8 + j];
        }
        // ---- B fragments: load once, reuse across both M tiles
        #pragma unroll
        for (int nt = 0; nt < 4; ++nt) {
            const int nloc = wn * 64 + nt * 16 + l15;
            v16bf bf;
            #pragma unroll
            for (int j = 0; j < 16; ++j) bf[j] = Bs[nloc][half * 16 + j];
            #pragma unroll
            for (int mt = 0; mt < 2; ++mt)
                acc[mt][nt] = __builtin_amdgcn_wmma_f32_16x16x32_bf16(
                    false, af[mt], false, bf, (short)0, acc[mt][nt], false, false);
        }
        __syncthreads();
    }

    // ---- epilogue: + b_enc, store preact (C/D ISA layout)
    #pragma unroll
    for (int nt = 0; nt < 4; ++nt) {
        const int n  = n0 + wn * 64 + nt * 16 + l15;
        const float be = b_enc[n];
        #pragma unroll
        for (int mt = 0; mt < 2; ++mt) {
            const int mb = m0 + wm * 32 + mt * 16 + half * 8;
            #pragma unroll
            for (int r = 0; r < 8; ++r)
                preact[(size_t)(mb + r) * L_LAT + n] = acc[mt][nt][r] + be;
        }
    }
}

// ---------------------------------------------------------------------------
// Kernel 2: per-row top-32 (iterative argmax, owner-only rescan).
// Row is pulled into LDS with GLOBAL_LOAD_ASYNC_TO_LDS_B128 (ASYNCcnt path);
// latents row is zero-filled + ReLU-scattered; (val,idx) stashed for decode.
// ---------------------------------------------------------------------------
__global__ __launch_bounds__(256) void sae_topk_scatter(
    const float* __restrict__ preact, float* __restrict__ latents,
    float* __restrict__ wvals, int* __restrict__ widx)
{
    extern __shared__ unsigned char smem[];
    float* s_row = (float*)smem;                                // 32768 f32
    float* s_rv  = (float*)(smem + L_LAT * 4);                  // 256 f32
    int*   s_ri  = (int*)  (smem + L_LAT * 4 + 256 * 4);        // 256 i32
    float* s_tv  = (float*)(smem + L_LAT * 4 + 256 * 8);        // 32 f32
    int*   s_ti  = (int*)  (smem + L_LAT * 4 + 256 * 8 + 128);  // 32 i32

    const int b   = blockIdx.x;
    const int tid = threadIdx.x;
    const float NEG = -3.402823466e38f;

    const float4* row4  = (const float4*)(preact + (size_t)b * L_LAT);
    float4*       lrow4 = (float4*)(latents + (size_t)b * L_LAT);
    float4 z4; z4.x = z4.y = z4.z = z4.w = 0.0f;

    // Async DMA the preact row into LDS (16B per lane per issue, 32 issues).
    const unsigned lds_base = (unsigned)(size_t)s_row;   // low 32b = LDS offset
    for (int i = tid; i < L_LAT / 4; i += 256) {
        const float4* gp = row4 + i;
        unsigned loff = lds_base + (unsigned)i * 16u;
        asm volatile("global_load_async_to_lds_b128 %0, %1, off"
                     :: "v"(loff), "v"(gp) : "memory");
        lrow4[i] = z4;                                   // zero latents row
    }
    asm volatile("s_wait_asynccnt 0x0" ::: "memory");
    __syncthreads();

    // per-thread local argmax over its 128-element chunk (lowest index wins)
    const int base = tid << 7;
    float bv = NEG; int bi = base;
    for (int j = 0; j < 128; ++j) {
        float v = s_row[base + j];
        if (v > bv) { bv = v; bi = base + j; }
    }

    for (int it = 0; it < TOPK; ++it) {
        s_rv[tid] = bv; s_ri[tid] = bi;
        __syncthreads();
        for (int s = 128; s > 0; s >>= 1) {              // 256 -> 1 reduce
            if (tid < s) {
                float v2 = s_rv[tid + s]; int i2 = s_ri[tid + s];
                float v1 = s_rv[tid];     int i1 = s_ri[tid];
                if (v2 > v1 || (v2 == v1 && i2 < i1)) { s_rv[tid] = v2; s_ri[tid] = i2; }
            }
            __syncthreads();
        }
        const float wv = s_rv[0];
        const int   wi = s_ri[0];
        if (tid == 0) { s_tv[it] = wv; s_ti[it] = wi; }
        if ((wi >> 7) == tid) {                          // owner masks + rescans
            s_row[wi] = NEG;
            bv = NEG; bi = base;
            for (int j = 0; j < 128; ++j) {
                float v = s_row[base + j];
                if (v > bv) { bv = v; bi = base + j; }
            }
        }
        __syncthreads();
    }

    if (tid < TOPK) {
        float v = s_tv[tid];
        v = v > 0.0f ? v : 0.0f;                         // ReLU
        int ix = s_ti[tid];
        wvals[b * TOPK + tid] = v;
        widx [b * TOPK + tid] = ix;
        latents[(size_t)b * L_LAT + ix] = v;             // scatter
    }
}

// ---------------------------------------------------------------------------
// Kernel 3: sparse decode  out[b,:] = b_dec + sum_k val[k] * W_dec[idx[k],:]
// ---------------------------------------------------------------------------
__global__ __launch_bounds__(256) void sae_decode(
    const float* __restrict__ wvals, const int* __restrict__ widx,
    const float* __restrict__ W_dec, const float* __restrict__ b_dec,
    float* __restrict__ out, float* __restrict__ dead_slot)
{
    __shared__ float sv[TOPK];
    __shared__ int   si[TOPK];
    const int b   = blockIdx.x;
    const int tid = threadIdx.x;

    if (tid < TOPK) {
        sv[tid] = wvals[b * TOPK + tid];
        si[tid] = widx [b * TOPK + tid];
    }
    __syncthreads();

    const int d = tid * 4;
    if (tid < TOPK)   // warm the gathered rows (global_prefetch_b8)
        __builtin_prefetch(W_dec + (size_t)si[tid] * D_IN, 0, 0);

    float4 acc = *(const float4*)(b_dec + d);
    #pragma unroll 8
    for (int k = 0; k < TOPK; ++k) {
        const float4 w = *(const float4*)(W_dec + (size_t)si[k] * D_IN + d);
        const float v = sv[k];
        acc.x += v * w.x; acc.y += v * w.y; acc.z += v * w.z; acc.w += v * w.w;
    }
    *(float4*)(out + (size_t)b * D_IN + d) = acc;

    if (b == 0 && tid == 0) ((int*)dead_slot)[0] = 0;   // num_dead = 0
}

// ---------------------------------------------------------------------------
extern "C" void kernel_launch(void* const* d_in, const int* in_sizes, int n_in,
                              void* d_out, int out_size, void* d_ws, size_t ws_size,
                              hipStream_t stream) {
    const float* x     = (const float*)d_in[0];
    const float* W_enc = (const float*)d_in[1];
    const float* b_enc = (const float*)d_in[2];
    const float* W_dec = (const float*)d_in[3];
    const float* b_dec = (const float*)d_in[4];

    float* out_f   = (float*)d_out;
    float* latents = out_f;                                      // [B, L]
    float* outp    = latents + (size_t)B_TOK * L_LAT;            // [B, D]
    float* preact  = outp    + (size_t)B_TOK * D_IN;             // [B, L]
    float* dead    = preact  + (size_t)B_TOK * L_LAT;            // [1]

    float* wvals = (float*)d_ws;                                 // [B, 32]
    int*   widxp = (int*)((char*)d_ws + (size_t)B_TOK * TOPK * sizeof(float));

    dim3 gridE(L_LAT / TN, B_TOK / TM);   // (128, 64)
    sae_encode_wmma<<<gridE, 256, 0, stream>>>(x, W_enc, b_enc, preact);

    const size_t smem = (size_t)L_LAT * 4 + 256 * 8 + TOPK * 8;  // 133376 B
    sae_topk_scatter<<<B_TOK, 256, smem, stream>>>(preact, latents, wvals, widxp);

    sae_decode<<<B_TOK, 256, 0, stream>>>(wvals, widxp, W_dec, b_dec, outp, dead);
}
